// ResidualLinearAttention_23888608100505
// MI455X (gfx1250) — matrix operand order repaired and checked by reading, not compiled, not run on hardware
//
#include <hip/hip_runtime.h>
#include <cstdint>

typedef __attribute__((ext_vector_type(16))) _Float16 v16h;
typedef __attribute__((ext_vector_type(8)))  _Float16 v8h;
typedef __attribute__((ext_vector_type(8)))  float    v8f;
typedef __attribute__((ext_vector_type(4)))  _Float16 h4;
typedef _Float16 h16;

#define NB     16          // batch
#define CIN    256
#define NPIX   4096        // 64*64
#define NHEAD  8
#define DHEAD  64
#define HIDDEN 512
#define QROWS  1536        // 3*HIDDEN
#define EPSV   1e-5f

// ---------------- WMMA fragment helpers (CDNA5 ISA 7.12.2 layouts) ----------
// Both LDS tiles are stored K-contiguous so fragments are vector ds loads:
//   A 16x32 f16: lane(0..15)=row M, half=lane>>4; elements = two contiguous
//                8-half runs at K = {8*half..} and {16+8*half..}.
//   B 32x16 f16: lane(0..15)=col N; elements = two contiguous 8-half runs at
//                K = {8*half..} and {16+8*half..}.
__device__ __forceinline__ v16h frag_cat(v8h lo, v8h hi) {
    return __builtin_shufflevector(lo, hi, 0,1,2,3,4,5,6,7,8,9,10,11,12,13,14,15);
}
__device__ __forceinline__ v16h load_fragA(const h16* As, int ld, int row, int kk, int half) {
    const h16* p = As + row * ld + kk + (half << 3);
    return frag_cat(*(const v8h*)p, *(const v8h*)(p + 16));
}
__device__ __forceinline__ v16h load_fragB(const h16* BsT, int ld, int col, int half) {
    const h16* p = BsT + col * ld + (half << 3);
    return frag_cat(*(const v8h*)p, *(const v8h*)(p + 16));
}
__device__ __forceinline__ v8f wmma16(v16h a, v16h b, v8f c) {
    return __builtin_amdgcn_wmma_f32_16x16x32_f16(false, a, false, b, (short)0, c, false, false);
}

// ---------------- K0: f32 -> f16 convert ------------------------------------
__global__ void cvt_f16_kernel(const float* __restrict__ src, h16* __restrict__ dst, int n) {
    int i = blockIdx.x * 256 + threadIdx.x;
    if (i < n) dst[i] = (h16)src[i];
}

// ---------------- K1: GroupNorm1 stats (32 groups) --------------------------
__global__ void gn1_stats_kernel(const float* __restrict__ x, float* __restrict__ stat) {
    const int bg = blockIdx.x;                      // b*32+g, group = contiguous 32768 f32
    const float4* p = (const float4*)(x + (size_t)bg * 32768);
    float s = 0.f, sq = 0.f;
    for (int i = threadIdx.x; i < 8192; i += 256) {
        float4 v = p[i];
        s  += v.x + v.y + v.z + v.w;
        sq += v.x * v.x + v.y * v.y + v.z * v.z + v.w * v.w;
    }
    __shared__ float ss[256], sv[256];
    ss[threadIdx.x] = s; sv[threadIdx.x] = sq;
    __syncthreads();
    for (int off = 128; off > 0; off >>= 1) {
        if (threadIdx.x < off) { ss[threadIdx.x] += ss[threadIdx.x + off]; sv[threadIdx.x] += sv[threadIdx.x + off]; }
        __syncthreads();
    }
    if (threadIdx.x == 0) {
        float mean = ss[0] * (1.f / 32768.f);
        float var  = sv[0] * (1.f / 32768.f) - mean * mean;
        stat[bg * 2 + 0] = mean;
        stat[bg * 2 + 1] = rsqrtf(var + EPSV);
    }
}

// ---------------- K2: GroupNorm1 apply -> f16 -------------------------------
__global__ void gn1_apply_kernel(const float* __restrict__ x, const float* __restrict__ stat,
                                 const float* __restrict__ w, const float* __restrict__ bb,
                                 h16* __restrict__ xn) {
    int i4 = blockIdx.x * 256 + threadIdx.x;        // float4 index, total 4194304
    int i  = i4 << 2;
    int c  = (i >> 12) & 255;
    int bg = i >> 15;                               // b*32 + g
    float mu = stat[bg * 2], rs = stat[bg * 2 + 1];
    float sw = w[c] * rs, sb = bb[c] - mu * sw;
    float4 v = ((const float4*)x)[i4];
    h4 o;
    o[0] = (h16)(v.x * sw + sb); o[1] = (h16)(v.y * sw + sb);
    o[2] = (h16)(v.z * sw + sb); o[3] = (h16)(v.w * sw + sb);
    *(h4*)(xn + i) = o;
}

// ---------------- K3: QKV GEMM (M=1536,K=256,N=4096 per batch) --------------
// 128x64 block tile, 8 waves as 4x2 grid of 32x32 wave tiles (4 wmma each),
// double-buffered LDS: one barrier per K-step.
__global__ __launch_bounds__(256) void qkv_gemm_kernel(
    const h16* __restrict__ Wq, const h16* __restrict__ XN,
    const float* __restrict__ qkvb, h16* __restrict__ QKV) {
    const int b   = blockIdx.z;
    const int gm0 = blockIdx.y * 128;
    const int gn0 = blockIdx.x * 64;
    __shared__ h16 As[2][128 * 40];    // As[m][k]
    __shared__ h16 BsT[2][64 * 40];    // BsT[n][k]
    const int tid = threadIdx.x, lane = tid & 31, wave = tid >> 5;
    const int half = lane >> 4, idx = lane & 15;
    const int wm0 = (wave & 3) << 5, wn0 = (wave >> 2) << 5;
    const h16* Ag = Wq + (size_t)gm0 * 256;
    const h16* Bg = XN + (size_t)b * CIN * NPIX + gn0;
    const int ar = tid >> 2, ac8 = (tid & 3) << 3;     // A: 2 uint4 per thread
    const int bk = tid >> 3, bn8 = (tid & 7) << 3;     // B: 1 v8h per thread

    uint4 pa0, pa1; v8h pb;
    auto fetch = [&](int kk) {
        pa0 = *(const uint4*)(Ag + (size_t)ar * 256 + kk + ac8);
        pa1 = *(const uint4*)(Ag + (size_t)(ar + 64) * 256 + kk + ac8);
        pb  = *(const v8h*)(Bg + (size_t)(kk + bk) * NPIX + bn8);
    };
    auto stage = [&](int buf) {
        *(uint4*)(&As[buf][ar * 40 + ac8])        = pa0;
        *(uint4*)(&As[buf][(ar + 64) * 40 + ac8]) = pa1;
#pragma unroll
        for (int i = 0; i < 8; ++i) BsT[buf][(bn8 + i) * 40 + bk] = pb[i];
    };

    v8f acc00 = {}, acc01 = {}, acc10 = {}, acc11 = {};
    fetch(0); stage(0); __syncthreads();
    for (int step = 0; step < 8; ++step) {
        const int cur = step & 1;
        if (step < 7) { fetch((step + 1) << 5); stage(cur ^ 1); }
        v16h a0 = load_fragA(As[cur], 40, wm0 + idx, 0, half);
        v16h a1 = load_fragA(As[cur], 40, wm0 + 16 + idx, 0, half);
        v16h b0 = load_fragB(BsT[cur], 40, wn0 + idx, half);
        v16h b1 = load_fragB(BsT[cur], 40, wn0 + 16 + idx, half);
        acc00 = wmma16(a0, b0, acc00);
        acc01 = wmma16(a0, b1, acc01);
        acc10 = wmma16(a1, b0, acc10);
        acc11 = wmma16(a1, b1, acc11);
        __syncthreads();
    }
#pragma unroll
    for (int ai = 0; ai < 2; ++ai) {
        v8f accA = ai ? acc10 : acc00;
        v8f accB = ai ? acc11 : acc01;
#pragma unroll
        for (int r = 0; r < 8; ++r) {
            int o = gm0 + wm0 + ai * 16 + r + (half << 3);
            float bias  = qkvb[o];
            float scale = (o < HIDDEN) ? 0.125f : 1.0f;   // fold q *= DIM_HEAD^-0.5
            h16* out = QKV + ((size_t)b * QROWS + o) * NPIX + gn0 + wn0 + idx;
            out[0]  = (h16)((accA[r] + bias) * scale);
            out[16] = (h16)((accB[r] + bias) * scale);
        }
    }
}

// ---------------- K4: softmax over n (rows of q), in place ------------------
__global__ void q_softmax_kernel(h16* __restrict__ QKV) {
    const int b = blockIdx.x >> 9, o = blockIdx.x & 511;
    h16* row = QKV + ((size_t)b * QROWS + o) * NPIX;
    __shared__ float red[256];
    float m = -1e30f;
    for (int i = threadIdx.x; i < NPIX; i += 256) m = fmaxf(m, (float)row[i]);
    red[threadIdx.x] = m; __syncthreads();
    for (int off = 128; off > 0; off >>= 1) {
        if (threadIdx.x < off) red[threadIdx.x] = fmaxf(red[threadIdx.x], red[threadIdx.x + off]);
        __syncthreads();
    }
    m = red[0]; __syncthreads();
    float s = 0.f;
    for (int i = threadIdx.x; i < NPIX; i += 256) s += __expf((float)row[i] - m);
    red[threadIdx.x] = s; __syncthreads();
    for (int off = 128; off > 0; off >>= 1) {
        if (threadIdx.x < off) red[threadIdx.x] += red[threadIdx.x + off];
        __syncthreads();
    }
    float inv = 1.f / red[0];
    for (int i = threadIdx.x; i < NPIX; i += 256)
        row[i] = (h16)(__expf((float)row[i] - m) * inv);
}

// ---------------- K5: softmax over d=64 (columns of k), in place ------------
__global__ void k_softmax_kernel(h16* __restrict__ QKV) {
    int t = blockIdx.x * 256 + threadIdx.x;   // 16*8*4096 threads
    int n = t & 4095, h = (t >> 12) & 7, b = t >> 15;
    h16* col = QKV + ((size_t)b * QROWS + HIDDEN + h * DHEAD) * NPIX + n;
    float m = -1e30f;
    for (int d = 0; d < 64; ++d) m = fmaxf(m, (float)col[(size_t)d * NPIX]);
    float s = 0.f;
    for (int d = 0; d < 64; ++d) s += __expf((float)col[(size_t)d * NPIX] - m);
    float inv = 1.f / s;
    for (int d = 0; d < 64; ++d) {
        size_t o = (size_t)d * NPIX;
        col[o] = (h16)(__expf((float)col[o] - m) * inv);
    }
}

// ---------------- K6: ctx^T[e][d] = (k_norm (64xN) * v^T (Nx64))^T ----------
// double-buffered (K loop = 128 steps), all-vector staging.
__global__ __launch_bounds__(256) void ctx_gemm_kernel(const h16* __restrict__ QKV,
                                                       h16* __restrict__ CTXT) {
    const int bh = blockIdx.x, b = bh >> 3, h = bh & 7;
    const h16* Kn = QKV + ((size_t)b * QROWS + HIDDEN + h * DHEAD) * NPIX;
    const h16* V  = QKV + ((size_t)b * QROWS + 2 * HIDDEN + h * DHEAD) * NPIX;
    __shared__ h16 As[2][64 * 40];     // As[d][k=n]
    __shared__ h16 BsT[2][64 * 40];    // BsT[e][k=n]  (v is naturally [e][n])
    const int tid = threadIdx.x, lane = tid & 31, wave = tid >> 5;
    const int half = lane >> 4, idx = lane & 15;
    const int wm0 = (wave & 3) << 4, wn0 = (wave >> 2) << 5;
    const int r = tid >> 2, c8 = (tid & 3) << 3;

    uint4 pa, pb;
    auto fetch = [&](int kk) {
        pa = *(const uint4*)(Kn + (size_t)r * NPIX + kk + c8);
        pb = *(const uint4*)(V  + (size_t)r * NPIX + kk + c8);
    };
    auto stage = [&](int buf) {
        *(uint4*)(&As[buf][r * 40 + c8])  = pa;
        *(uint4*)(&BsT[buf][r * 40 + c8]) = pb;
    };

    v8f acc0 = {}, acc1 = {};
    fetch(0); stage(0); __syncthreads();
    for (int step = 0; step < 128; ++step) {
        const int cur = step & 1;
        if (step < 127) { fetch((step + 1) << 5); stage(cur ^ 1); }
        v16h a  = load_fragA(As[cur], 40, wm0 + idx, 0, half);
        v16h b0 = load_fragB(BsT[cur], 40, wn0 + idx, half);
        v16h b1 = load_fragB(BsT[cur], 40, wn0 + 16 + idx, half);
        acc0 = wmma16(a, b0, acc0);
        acc1 = wmma16(a, b1, acc1);
        __syncthreads();
    }
    h16* out = CTXT + (size_t)bh * DHEAD * DHEAD;   // write TRANSPOSED: [e][d]
#pragma unroll
    for (int rr = 0; rr < 8; ++rr) {
        int d = wm0 + rr + (half << 3);
        out[(wn0 + idx) * 64 + d]      = (h16)acc0[rr];
        out[(wn0 + 16 + idx) * 64 + d] = (h16)acc1[rr];
    }
}

// ---------------- K7: att = ctx^T (64x64) * q_norm (64xN) per (b,h) ---------
__global__ __launch_bounds__(256) void att_gemm_kernel(const h16* __restrict__ QKV,
                                                       const h16* __restrict__ CTXT,
                                                       h16* __restrict__ ATT) {
    const int bh = blockIdx.y, b = bh >> 3, h = bh & 7;
    const int gn0 = blockIdx.x * 64;
    const h16* Q = QKV + ((size_t)b * QROWS + h * DHEAD) * NPIX;
    const h16* C = CTXT + (size_t)bh * DHEAD * DHEAD;   // [e][d], K-contiguous
    __shared__ h16 As[64 * 72];     // As[e][k=d], both k-steps staged once
    __shared__ h16 BsT[64 * 40];    // BsT[n][k=d]
    const int tid = threadIdx.x, lane = tid & 31, wave = tid >> 5;
    const int half = lane >> 4, idx = lane & 15;
    const int wm0 = (wave & 3) << 4, wn0 = (wave >> 2) << 5;
    { int r = tid & 63, c16 = (tid >> 6) << 4;          // vector stage of ctx^T
      *(uint4*)(&As[r * 72 + c16])     = *(const uint4*)(C + (size_t)r * 64 + c16);
      *(uint4*)(&As[r * 72 + c16 + 8]) = *(const uint4*)(C + (size_t)r * 64 + c16 + 8); }
    v8f acc0 = {}, acc1 = {};
#pragma unroll
    for (int kk = 0; kk < 64; kk += 32) {
        { int k = tid >> 3, n8 = (tid & 7) << 3;        // q: coalesced read + scatter
          v8h v = *(const v8h*)(Q + (size_t)(kk + k) * NPIX + gn0 + n8);
#pragma unroll
          for (int i = 0; i < 8; ++i) BsT[(n8 + i) * 40 + k] = v[i]; }
        __syncthreads();
        v16h a  = load_fragA(As, 72, wm0 + idx, kk, half);
        v16h b0 = load_fragB(BsT, 40, wn0 + idx, half);
        v16h b1 = load_fragB(BsT, 40, wn0 + 16 + idx, half);
        acc0 = wmma16(a, b0, acc0);
        acc1 = wmma16(a, b1, acc1);
        __syncthreads();
    }
#pragma unroll
    for (int r = 0; r < 8; ++r) {
        int e = wm0 + r + (half << 3);
        int c = h * DHEAD + e;
        h16* out = ATT + ((size_t)b * HIDDEN + c) * NPIX;
        out[gn0 + wn0 + idx]      = (h16)acc0[r];
        out[gn0 + wn0 + 16 + idx] = (h16)acc1[r];
    }
}

// ---------------- K8: out projection (M=256,K=512,N=4096 per batch) ---------
// 128x64 block tile, 32x32 wave tiles, double-buffered LDS.
__global__ __launch_bounds__(256) void proj_gemm_kernel(
    const h16* __restrict__ Wo, const h16* __restrict__ ATT,
    const float* __restrict__ ob, float* __restrict__ O2) {
    const int b   = blockIdx.z;
    const int gm0 = blockIdx.y * 128;
    const int gn0 = blockIdx.x * 64;
    __shared__ h16 As[2][128 * 40];    // As[o][k=c]
    __shared__ h16 BsT[2][64 * 40];    // BsT[n][k=c]
    const int tid = threadIdx.x, lane = tid & 31, wave = tid >> 5;
    const int half = lane >> 4, idx = lane & 15;
    const int wm0 = (wave & 3) << 5, wn0 = (wave >> 2) << 5;
    const h16* Ag = Wo + (size_t)gm0 * 512;
    const h16* Bg = ATT + (size_t)b * HIDDEN * NPIX + gn0;
    const int ar = tid >> 2, ac8 = (tid & 3) << 3;
    const int bk = tid >> 3, bn8 = (tid & 7) << 3;

    uint4 pa0, pa1; v8h pb;
    auto fetch = [&](int kk) {
        pa0 = *(const uint4*)(Ag + (size_t)ar * 512 + kk + ac8);
        pa1 = *(const uint4*)(Ag + (size_t)(ar + 64) * 512 + kk + ac8);
        pb  = *(const v8h*)(Bg + (size_t)(kk + bk) * NPIX + bn8);
    };
    auto stage = [&](int buf) {
        *(uint4*)(&As[buf][ar * 40 + ac8])        = pa0;
        *(uint4*)(&As[buf][(ar + 64) * 40 + ac8]) = pa1;
#pragma unroll
        for (int i = 0; i < 8; ++i) BsT[buf][(bn8 + i) * 40 + bk] = pb[i];
    };

    v8f acc00 = {}, acc01 = {}, acc10 = {}, acc11 = {};
    fetch(0); stage(0); __syncthreads();
    for (int step = 0; step < 16; ++step) {
        const int cur = step & 1;
        if (step < 15) { fetch((step + 1) << 5); stage(cur ^ 1); }
        v16h a0 = load_fragA(As[cur], 40, wm0 + idx, 0, half);
        v16h a1 = load_fragA(As[cur], 40, wm0 + 16 + idx, 0, half);
        v16h b0 = load_fragB(BsT[cur], 40, wn0 + idx, half);
        v16h b1 = load_fragB(BsT[cur], 40, wn0 + 16 + idx, half);
        acc00 = wmma16(a0, b0, acc00);
        acc01 = wmma16(a0, b1, acc01);
        acc10 = wmma16(a1, b0, acc10);
        acc11 = wmma16(a1, b1, acc11);
        __syncthreads();
    }
#pragma unroll
    for (int ai = 0; ai < 2; ++ai) {
        v8f accA = ai ? acc10 : acc00;
        v8f accB = ai ? acc11 : acc01;
#pragma unroll
        for (int r = 0; r < 8; ++r) {
            int o = gm0 + wm0 + ai * 16 + r + (half << 3);
            float bias = ob[o];
            float* out = O2 + ((size_t)b * CIN + o) * NPIX + gn0 + wn0 + idx;
            out[0]  = accA[r] + bias;
            out[16] = accB[r] + bias;
        }
    }
}

// ---------------- K9/K10: GroupNorm2 (1 group = layer norm per batch) -------
__global__ void gn2_partial_kernel(const float* __restrict__ O2, float* __restrict__ part) {
    const float4* p = (const float4*)(O2 + (size_t)blockIdx.y * 1048576 + (size_t)blockIdx.x * 16384);
    float s = 0.f, sq = 0.f;
    for (int i = threadIdx.x; i < 4096; i += 256) {
        float4 v = p[i];
        s  += v.x + v.y + v.z + v.w;
        sq += v.x * v.x + v.y * v.y + v.z * v.z + v.w * v.w;
    }
    __shared__ float ss[256], sv[256];
    ss[threadIdx.x] = s; sv[threadIdx.x] = sq;
    __syncthreads();
    for (int off = 128; off > 0; off >>= 1) {
        if (threadIdx.x < off) { ss[threadIdx.x] += ss[threadIdx.x + off]; sv[threadIdx.x] += sv[threadIdx.x + off]; }
        __syncthreads();
    }
    if (threadIdx.x == 0) {
        int i = blockIdx.y * 64 + blockIdx.x;
        part[i * 2 + 0] = ss[0];
        part[i * 2 + 1] = sv[0];
    }
}

__global__ void gn2_final_kernel(const float* __restrict__ part, float* __restrict__ stat) {
    const int b = blockIdx.x, t = threadIdx.x;   // 64 threads
    __shared__ float ss[64], sv[64];
    ss[t] = part[(b * 64 + t) * 2 + 0];
    sv[t] = part[(b * 64 + t) * 2 + 1];
    __syncthreads();
    for (int off = 32; off > 0; off >>= 1) {
        if (t < off) { ss[t] += ss[t + off]; sv[t] += sv[t + off]; }
        __syncthreads();
    }
    if (t == 0) {
        float mean = ss[0] * (1.f / 1048576.f);
        float var  = sv[0] * (1.f / 1048576.f) - mean * mean;
        stat[b * 2 + 0] = mean;
        stat[b * 2 + 1] = rsqrtf(var + EPSV);
    }
}

__global__ void gn2_apply_kernel(const float* __restrict__ O2, const float* __restrict__ stat,
                                 const float* __restrict__ w, const float* __restrict__ bb,
                                 const float* __restrict__ x, float* __restrict__ out) {
    int i4 = blockIdx.x * 256 + threadIdx.x;   // total 4194304 float4
    int i  = i4 << 2;
    int b  = i >> 20;
    int c  = (i >> 12) & 255;
    float mu = stat[b * 2], rs = stat[b * 2 + 1];
    float sw = w[c] * rs, sb = bb[c] - mu * sw;
    float4 v = ((const float4*)O2)[i4];
    float4 r = ((const float4*)x)[i4];
    float4 o;
    o.x = v.x * sw + sb + r.x; o.y = v.y * sw + sb + r.y;
    o.z = v.z * sw + sb + r.z; o.w = v.w * sw + sb + r.w;
    ((float4*)out)[i4] = o;
}

// ---------------------------------------------------------------------------
static inline size_t alignup(size_t x) { return (x + 255) & ~(size_t)255; }

extern "C" void kernel_launch(void* const* d_in, const int* in_sizes, int n_in,
                              void* d_out, int out_size, void* d_ws, size_t ws_size,
                              hipStream_t stream) {
    const float* x     = (const float*)d_in[0];
    const float* gn1w  = (const float*)d_in[1];
    const float* gn1b  = (const float*)d_in[2];
    const float* qkvw  = (const float*)d_in[3];
    const float* qkvb  = (const float*)d_in[4];
    const float* outw  = (const float*)d_in[5];
    const float* outb  = (const float*)d_in[6];
    const float* gn2w  = (const float*)d_in[7];
    const float* gn2b  = (const float*)d_in[8];
    float* out = (float*)d_out;

    // workspace layout (~304 MB)
    char* ws = (char*)d_ws;
    size_t off = 0;
    h16*   Wq   = (h16*)(ws + off);  off = alignup(off + (size_t)QROWS * CIN * 2);
    h16*   Wo   = (h16*)(ws + off);  off = alignup(off + (size_t)CIN * HIDDEN * 2);
    float* g1   = (float*)(ws + off); off = alignup(off + 512 * 2 * 4);
    float* part = (float*)(ws + off); off = alignup(off + 1024 * 2 * 4);
    float* g2   = (float*)(ws + off); off = alignup(off + 32 * 4);
    h16*   CTXT = (h16*)(ws + off);  off = alignup(off + (size_t)NB * NHEAD * DHEAD * DHEAD * 2);
    h16*   XN   = (h16*)(ws + off);  off = alignup(off + (size_t)NB * CIN * NPIX * 2);
    h16*   QKV  = (h16*)(ws + off);  off = alignup(off + (size_t)NB * QROWS * NPIX * 2);
    h16*   ATT  = (h16*)(ws + off);  off = alignup(off + (size_t)NB * HIDDEN * NPIX * 2);
    float* O2   = (float*)QKV;       // aliases qkv buffer (dead after att_gemm)

    cvt_f16_kernel<<<dim3(1536), 256, 0, stream>>>(qkvw, Wq, QROWS * CIN);
    cvt_f16_kernel<<<dim3(512),  256, 0, stream>>>(outw, Wo, CIN * HIDDEN);
    gn1_stats_kernel<<<dim3(512), 256, 0, stream>>>(x, g1);
    gn1_apply_kernel<<<dim3(16384), 256, 0, stream>>>(x, g1, gn1w, gn1b, XN);
    qkv_gemm_kernel<<<dim3(64, 12, 16), 256, 0, stream>>>(Wq, XN, qkvb, QKV);
    q_softmax_kernel<<<dim3(NB * HIDDEN), 256, 0, stream>>>(QKV);
    k_softmax_kernel<<<dim3(2048), 256, 0, stream>>>(QKV);
    ctx_gemm_kernel<<<dim3(NB * NHEAD), 256, 0, stream>>>(QKV, CTXT);
    att_gemm_kernel<<<dim3(64, NB * NHEAD), 256, 0, stream>>>(QKV, CTXT, ATT);
    proj_gemm_kernel<<<dim3(64, 2, 16), 256, 0, stream>>>(Wo, ATT, outb, O2);
    gn2_partial_kernel<<<dim3(64, 16), 256, 0, stream>>>(O2, part);
    gn2_final_kernel<<<dim3(16), 64, 0, stream>>>(part, g2);
    gn2_apply_kernel<<<dim3(16384), 256, 0, stream>>>(O2, g2, gn2w, gn2b, x, out);
}
